// HierarchicalMemory_80590766342680
// MI455X (gfx1250) — matrix operand order repaired and checked by reading, not compiled, not run on hardware
//
#include <hip/hip_runtime.h>
#include <hip/hip_bf16.h>
#include <math.h>

// ---- CDNA5 WMMA vector types ----
typedef __attribute__((ext_vector_type(16))) __bf16 v16bf;
typedef __attribute__((ext_vector_type(8)))  __bf16 bf16x8;
typedef __attribute__((ext_vector_type(8)))  float  v8f;

#define TILE_M 128
#define TILE_N 128
#define TILE_K 32
#define LDS_STRIDE 40   // 32 + 8 bf16 padding to break bank conflicts

union V16U { v16bf v; bf16x8 h[2]; };

// ---- bit-level bf16 conversions (round-to-nearest-even), backend-safe ----
__device__ __forceinline__ __bf16 f2bf(float f) {
    unsigned u = __builtin_bit_cast(unsigned, f);
    unsigned r = (u + 0x7fffu + ((u >> 16) & 1u)) >> 16;
    unsigned short s = (unsigned short)r;
    return __builtin_bit_cast(__bf16, s);
}
__device__ __forceinline__ float bf2f(__bf16 h) {
    unsigned short s = __builtin_bit_cast(unsigned short, h);
    unsigned u = (unsigned)s << 16;
    return __builtin_bit_cast(float, u);
}

// ---- global -> register fragment fetch (16 bf16 per thread) ----
__device__ __forceinline__ void fetch16(const float* g, bf16x8* r) {
    const float4* g4 = (const float4*)g;
#pragma unroll
    for (int q = 0; q < 2; ++q) {
        float4 f0 = g4[q * 2 + 0];
        float4 f1 = g4[q * 2 + 1];
        bf16x8 h;
        h[0] = f2bf(f0.x); h[1] = f2bf(f0.y); h[2] = f2bf(f0.z); h[3] = f2bf(f0.w);
        h[4] = f2bf(f1.x); h[5] = f2bf(f1.y); h[6] = f2bf(f1.z); h[7] = f2bf(f1.w);
        r[q] = h;
    }
}
__device__ __forceinline__ void fetch16(const __bf16* g, bf16x8* r) {
    const bf16x8* g8 = (const bf16x8*)g;
    r[0] = g8[0];
    r[1] = g8[1];
}

__device__ __forceinline__ void commit16(__bf16* d, const bf16x8* r) {
    *(bf16x8*)(d)     = r[0];
    *(bf16x8*)(d + 8) = r[1];
}

// ---- output store ----
__device__ __forceinline__ void store_out(float* p, float v)  { *p = v; }
__device__ __forceinline__ void store_out(__bf16* p, float v) { *p = f2bf(v); }

// =====================================================================
// NT GEMM:  C[m,n] = scale * sum_k A[m,k] * B[n,k]  (+ Cadd[m,n] if HAS_ADD)
// A: [M,K] row-major (fp32 or bf16), B: [N,K] row-major (fp32 or bf16)
// C: [M,N] row-major (fp32 or bf16). M,N multiples of 128; K multiple of 32.
// 256 threads = 8 wave32; wave (wm,wn) computes a 32x64 sub-tile via
// 2x4 v_wmma_f32_16x16x32_bf16 tiles, double-buffered LDS staging.
// =====================================================================
template <typename AT, typename BT, typename OT, bool HAS_ADD>
__global__ __launch_bounds__(256)
void gemm_nt(const AT* __restrict__ A, const BT* __restrict__ B,
             OT* __restrict__ C, const float* __restrict__ Cadd,
             int M, int N, int K, float scale)
{
    __shared__ __bf16 ldsA[2][TILE_M * LDS_STRIDE];
    __shared__ __bf16 ldsB[2][TILE_N * LDS_STRIDE];

    const int tid  = threadIdx.x;
    const int lane = tid & 31;
    const int wave = tid >> 5;
    const int wm   = wave & 3;   // 0..3  -> 32-row strip
    const int wn   = wave >> 2;  // 0..1  -> 64-col strip
    const int lrow = lane & 15;
    const int lhi  = lane >> 4;

    const int bm = blockIdx.y * TILE_M;
    const int bn = blockIdx.x * TILE_N;

    // staging: each thread owns one 16-element chunk of the 128x32 tile
    const int ldrow = tid >> 1;         // 0..127
    const int ldcol = (tid & 1) * 16;   // 0 or 16

    const AT* gA = A + (size_t)(bm + ldrow) * (size_t)K + ldcol;
    const BT* gB = B + (size_t)(bn + ldrow) * (size_t)K + ldcol;

    bf16x8 ra[2], rb[2];

    // prologue: tile 0 -> buffer 0
    fetch16(gA, ra);
    fetch16(gB, rb);
    commit16(&ldsA[0][ldrow * LDS_STRIDE + ldcol], ra);
    commit16(&ldsB[0][ldrow * LDS_STRIDE + ldcol], rb);
    __syncthreads();

    v8f acc[2][4];
#pragma unroll
    for (int i = 0; i < 2; ++i)
#pragma unroll
        for (int j = 0; j < 4; ++j) {
            v8f z = {};
            acc[i][j] = z;
        }

    const int nk = K / TILE_K;
    for (int kt = 0; kt < nk; ++kt) {
        const int cur = kt & 1;

        if (kt + 2 < nk) {   // L2 prefetch two tiles ahead
            __builtin_prefetch(gA + (size_t)(kt + 2) * TILE_K, 0, 0);
            __builtin_prefetch(gB + (size_t)(kt + 2) * TILE_K, 0, 0);
        }
        if (kt + 1 < nk) {   // fetch next tile into registers (overlaps WMMA)
            fetch16(gA + (size_t)(kt + 1) * TILE_K, ra);
            fetch16(gB + (size_t)(kt + 1) * TILE_K, rb);
        }

        // ---- fragment loads, contiguous k order k = 16*lhi + s, same for A and B ----
        v16bf af[2];
#pragma unroll
        for (int i = 0; i < 2; ++i) {
            const __bf16* pa =
                &ldsA[cur][(wm * 32 + i * 16 + lrow) * LDS_STRIDE + lhi * 16];
            V16U u;
            u.h[0] = *(const bf16x8*)(pa);        // K = klo .. klo+7
            u.h[1] = *(const bf16x8*)(pa + 8);    // K = klo+8 .. klo+15
            af[i] = u.v;
        }
        v16bf bfr[4];
#pragma unroll
        for (int j = 0; j < 4; ++j) {
            const __bf16* pb =
                &ldsB[cur][(wn * 64 + j * 16 + lrow) * LDS_STRIDE + lhi * 16];
            V16U u;
            u.h[0] = *(const bf16x8*)(pb);        // K = klo .. klo+7
            u.h[1] = *(const bf16x8*)(pb + 8);    // K = klo+8 .. klo+15
            bfr[j] = u.v;
        }

#pragma unroll
        for (int i = 0; i < 2; ++i)
#pragma unroll
            for (int j = 0; j < 4; ++j)
                acc[i][j] = __builtin_amdgcn_wmma_f32_16x16x32_bf16(
                    false, af[i], false, bfr[j], (short)0, acc[i][j],
                    false, false);

        if (kt + 1 < nk) {
            const int nxt = cur ^ 1;
            commit16(&ldsA[nxt][ldrow * LDS_STRIDE + ldcol], ra);
            commit16(&ldsB[nxt][ldrow * LDS_STRIDE + ldcol], rb);
        }
        __syncthreads();
    }

    // ---- epilogue: C/D layout: VGPR r, lanes 0-15 -> M=r, lanes 16-31 -> M=r+8
#pragma unroll
    for (int i = 0; i < 2; ++i) {
        const int row0 = bm + wm * 32 + i * 16 + lhi * 8;
#pragma unroll
        for (int j = 0; j < 4; ++j) {
            const int col = bn + wn * 64 + j * 16 + lrow;
            const size_t base = (size_t)row0 * (size_t)N + col;

            float addv[8];
            if (HAS_ADD) {
                // issue all 8 loads up front -> single wait, no per-element stalls
#pragma unroll
                for (int r = 0; r < 8; ++r)
                    addv[r] = Cadd[base + (size_t)r * (size_t)N];
            }
#pragma unroll
            for (int r = 0; r < 8; ++r) {
                float v = acc[i][j][r] * scale;
                if (HAS_ADD) v += addv[r];
                store_out(&C[base + (size_t)r * (size_t)N], v);
            }
        }
    }
}

// =====================================================================
// Row softmax over bf16 matrix, in place. One 256-thread block per row.
// =====================================================================
__global__ __launch_bounds__(256)
void softmax_rows(__bf16* __restrict__ S, int ncols)
{
    const int tid = threadIdx.x;
    __bf16* p = S + (size_t)blockIdx.x * (size_t)ncols;
    __shared__ float red[256];

    float m = -3.0e38f;
    for (int c = tid; c < ncols; c += 256) m = fmaxf(m, bf2f(p[c]));
    red[tid] = m;
    __syncthreads();
#pragma unroll
    for (int s = 128; s > 0; s >>= 1) {
        if (tid < s) red[tid] = fmaxf(red[tid], red[tid + s]);
        __syncthreads();
    }
    m = red[0];
    __syncthreads();

    float sum = 0.0f;
    for (int c = tid; c < ncols; c += 256) sum += __expf(bf2f(p[c]) - m);
    red[tid] = sum;
    __syncthreads();
#pragma unroll
    for (int s = 128; s > 0; s >>= 1) {
        if (tid < s) red[tid] += red[tid + s];
        __syncthreads();
    }
    const float inv = 1.0f / red[0];
    __syncthreads();

    for (int c = tid; c < ncols; c += 256)
        p[c] = f2bf(__expf(bf2f(p[c]) - m) * inv);
}

// =====================================================================
// Launch: 8 NT GEMMs + 2 softmaxes cover the whole reference graph.
// =====================================================================
extern "C" void kernel_launch(void* const* d_in, const int* in_sizes, int n_in,
                              void* d_out, int out_size, void* d_ws, size_t ws_size,
                              hipStream_t stream)
{
    (void)in_sizes; (void)n_in; (void)out_size; (void)ws_size;

    const float* x   = (const float*)d_in[0];   // [B,L,D] = [2,2048,2048]
    const float* ek  = (const float*)d_in[1];   // [1,E,D]
    const float* ev  = (const float*)d_in[2];   // [1,E,D]
    const float* sem = (const float*)d_in[3];   // [M,D]
    const float* Wk  = (const float*)d_in[4];   // [D,D]
    const float* Wv  = (const float*)d_in[5];   // [D,D]
    float* out = (float*)d_out;                 // [memory_out | episodic_out]

    constexpr int D  = 2048;
    constexpr int E  = 16384;
    constexpr int SM = 2048;
    constexpr int BL = 2 * 2048;                // B*L = 4096

    // ---- workspace layout (bf16, ~302 MB) ----
    char* w = (char*)d_ws;
    __bf16* ke  = (__bf16*)w;  w += (size_t)E  * D * 2;   // ke  [E,D]
    __bf16* veT = (__bf16*)w;  w += (size_t)D  * E * 2;   // veT [D,E]
    __bf16* ks  = (__bf16*)w;  w += (size_t)SM * D * 2;   // ks  [M,D]
    __bf16* vsT = (__bf16*)w;  w += (size_t)D  * SM * 2;  // vsT [D,M]
    __bf16* Se  = (__bf16*)w;  w += (size_t)BL * E * 2;   // scores_e / P_e
    __bf16* Ss  = (__bf16*)w;  w += (size_t)BL * SM * 2;  // scores_s / P_s

    const dim3 blk(256);
    const float inv_sqrt_d = 1.0f / sqrtf((float)D);

    // --- projections (fp32 in -> bf16 out); V projections written transposed ---
    gemm_nt<float, float, __bf16, false><<<dim3(D / 128, E / 128), blk, 0, stream>>>(
        ek, Wk, ke, nullptr, E, D, D, 1.0f);                    // ke = ek @ Wk^T
    gemm_nt<float, float, __bf16, false><<<dim3(E / 128, D / 128), blk, 0, stream>>>(
        Wv, ev, veT, nullptr, D, E, D, 1.0f);                   // veT = Wv @ ev^T
    gemm_nt<float, float, __bf16, false><<<dim3(D / 128, SM / 128), blk, 0, stream>>>(
        sem, Wk, ks, nullptr, SM, D, D, 1.0f);                  // ks = sem @ Wk^T
    gemm_nt<float, float, __bf16, false><<<dim3(SM / 128, D / 128), blk, 0, stream>>>(
        Wv, sem, vsT, nullptr, D, SM, D, 1.0f);                 // vsT = Wv @ sem^T

    // --- attention scores ---
    gemm_nt<float, __bf16, __bf16, false><<<dim3(E / 128, BL / 128), blk, 0, stream>>>(
        x, ke, Se, nullptr, BL, E, D, 1.0f);                    // Se = x @ ke^T
    gemm_nt<float, __bf16, __bf16, false><<<dim3(SM / 128, BL / 128), blk, 0, stream>>>(
        x, ks, Ss, nullptr, BL, SM, D, inv_sqrt_d);             // Ss = x @ ks^T / sqrt(D)

    // --- softmax (in place) ---
    softmax_rows<<<BL, blk, 0, stream>>>(Se, E);
    softmax_rows<<<BL, blk, 0, stream>>>(Ss, SM);

    // --- outputs: episodic_out to d_out[BL*D..], memory_out (fused add) to d_out[0..] ---
    float* ep_out = out + (size_t)BL * D;
    gemm_nt<__bf16, __bf16, float, false><<<dim3(D / 128, BL / 128), blk, 0, stream>>>(
        Se, veT, ep_out, nullptr, BL, D, E, 1.0f);              // Oe = Pe @ veT^T
    gemm_nt<__bf16, __bf16, float, true><<<dim3(D / 128, BL / 128), blk, 0, stream>>>(
        Ss, vsT, out, ep_out, BL, D, SM, 1.0f);                 // Om = Ps @ vsT^T + Oe
}
